// ConsisGADGNNModule_64802466562525
// MI455X (gfx1250) — compile-verified, matrix operand order checked by reading, not run on hardware
//
#include <hip/hip_runtime.h>
#include <math.h>

typedef __attribute__((ext_vector_type(16))) _Float16 v16h;
typedef __attribute__((ext_vector_type(8)))  float    v8f;

// D = A(16x32 f16) x B(32x16 f16) + C(16x16 f32), wave32
__device__ __forceinline__ v8f wmma_f16(v16h a, v16h b, v8f c) {
  return __builtin_amdgcn_wmma_f32_16x16x32_f16(
      /*neg_a=*/false, a, /*neg_b=*/false, b,
      /*c_mod=*/(short)0, c, /*reuse_a=*/false, /*reuse_b=*/false);
}

__device__ __forceinline__ float elu1(float x) { return x > 0.f ? x : expm1f(x); }

// Build a 16x32 f16 A-fragment for this lane from a row of f32 data.
// ISA 16-bit A layout: element e -> K = (e&7) + 16*(e>>3) + 8*(lane>>4)
// p points at the 32-wide K-window of this lane's row (M = lane&15); hi = 8*(lane>>4)
__device__ __forceinline__ v16h a_frag_f32(const float* __restrict__ p, int hi) {
  v16h a;
  const float* p0 = p + hi;
  const float* p1 = p + 16 + hi;
#pragma unroll
  for (int i = 0; i < 8; ++i) {
    a[i]     = (_Float16)p0[i];
    a[i + 8] = (_Float16)p1[i];
  }
  return a;
}

__global__ void zero_ws_kernel(float* __restrict__ p, size_t n) {
  size_t i = (size_t)blockIdx.x * blockDim.x + threadIdx.x;
  size_t s = (size_t)gridDim.x * blockDim.x;
  for (; i < n; i += s) p[i] = 0.f;
}

__global__ void bn_finalize_kernel(const float* __restrict__ bsum,
                                   const float* __restrict__ bsq,
                                   const float* __restrict__ g,
                                   const float* __restrict__ b,
                                   float* __restrict__ scale,
                                   float* __restrict__ shift,
                                   int tc, float invE) {
  int i = blockIdx.x * blockDim.x + threadIdx.x;
  if (i >= tc) return;
  float mean = bsum[i] * invE;
  float var  = bsq[i] * invE - mean * mean;
  float s = g[i] * rsqrtf(var + 1e-5f);
  scale[i] = s;
  shift[i] = b[i] - mean * s;
}

// PASS 1: compute per-channel BN sums/sumsq over all edges of each type.
// PASS 2: recompute, apply BN affine, atomic scatter-add into nodeacc.
template <int PASS>
__launch_bounds__(256)
__global__ void edge_mlp_pass(
    const float* __restrict__ feat,
    const float* __restrict__ W1, const float* __restrict__ b1,
    const float* __restrict__ lng, const float* __restrict__ lnb,
    const float* __restrict__ W2, const float* __restrict__ b2,
    const int* __restrict__ esrc, const int* __restrict__ edst,
    float* __restrict__ bn_sum, float* __restrict__ bn_sumsq,
    const float* __restrict__ bn_scale, const float* __restrict__ bn_shift,
    float* __restrict__ nodeacc, int E) {
  // Pre-swizzled B fragments (weights), one 32B vector per (kstep, ntile, lane)
  __shared__ __align__(32) _Float16 sB1[4][4][32][16];  // 16 KB: W1 (K=128,N=64)
  __shared__ __align__(32) _Float16 sB2[2][4][32][16];  //  8 KB: W2 (K=64, N=64)
  __shared__ __align__(32) _Float16 sH[8][16 * 64];     // 16 KB: per-wave h tile
  __shared__ float sBn[128];                            // block BN accumulators

  const int t    = blockIdx.y;
  const int lane = threadIdx.x & 31;
  const int wave = threadIdx.x >> 5;
  const int lo   = lane & 15;
  const int hi   = (lane >> 4) << 3;  // 0 or 8

  const float* W1t = W1 + (size_t)t * 128 * 64;
  const float* W2t = W2 + (size_t)t * 64 * 64;

  for (int idx = threadIdx.x; idx < 4 * 4 * 32; idx += 256) {
    int ks = idx >> 7, nt = (idx >> 5) & 3, l = idx & 31;
    int n = nt * 16 + (l & 15);
    int kh = (l >> 4) << 3;
#pragma unroll
    for (int e = 0; e < 16; ++e) {
      int k = ks * 32 + (e & 7) + ((e >> 3) << 4) + kh;
      sB1[ks][nt][l][e] = (_Float16)W1t[k * 64 + n];
    }
  }
  for (int idx = threadIdx.x; idx < 2 * 4 * 32; idx += 256) {
    int ks = idx >> 7, nt = (idx >> 5) & 3, l = idx & 31;
    int n = nt * 16 + (l & 15);
    int kh = (l >> 4) << 3;
#pragma unroll
    for (int e = 0; e < 16; ++e) {
      int k = ks * 32 + (e & 7) + ((e >> 3) << 4) + kh;
      sB2[ks][nt][l][e] = (_Float16)W2t[k * 64 + n];
    }
  }
  if (PASS == 1 && threadIdx.x < 128) sBn[threadIdx.x] = 0.f;
  __syncthreads();

  // Per-lane channel parameters, channel n = nt*16 + lo
  float vb1[4], vg[4], vb[4], vb2[4], vscl[4], vsft[4];
#pragma unroll
  for (int nt = 0; nt < 4; ++nt) {
    int n = t * 64 + nt * 16 + lo;
    vb1[nt] = b1[n]; vg[nt] = lng[n]; vb[nt] = lnb[n]; vb2[nt] = b2[n];
    vscl[nt] = (PASS == 2) ? bn_scale[n] : 0.f;
    vsft[nt] = (PASS == 2) ? bn_shift[n] : 0.f;
  }

  float csum[4] = {0.f, 0.f, 0.f, 0.f}, csq[4] = {0.f, 0.f, 0.f, 0.f};

  const int* es = esrc + (size_t)t * E;
  const int* ed = edst + (size_t)t * E;
  const int ntiles = E >> 4;
  _Float16* hrow = sH[wave];

  for (int tile = blockIdx.x * 8 + wave; tile < ntiles; tile += gridDim.x << 3) {
    const int base = tile << 4;
    __builtin_prefetch(es + base + 2048, 0, 0);  // global_prefetch_b8
    const int sidx = es[base + lo];
    const int didx = ed[base + lo];
    const float* srow = feat + (size_t)sidx * 64;
    const float* drow = feat + (size_t)didx * 64;

    // ---- GEMM1: [16 x 128] x [128 x 64], x = concat(src, dst) ----
    v8f acc[4];
#pragma unroll
    for (int nt = 0; nt < 4; ++nt)
#pragma unroll
      for (int i = 0; i < 8; ++i) acc[nt][i] = 0.f;

#pragma unroll
    for (int ks = 0; ks < 4; ++ks) {
      const float* rp = (ks < 2) ? (srow + ks * 32) : (drow + (ks - 2) * 32);
      v16h a = a_frag_f32(rp, hi);
#pragma unroll
      for (int nt = 0; nt < 4; ++nt)
        acc[nt] = wmma_f16(a, *(const v16h*)sB1[ks][nt][lane], acc[nt]);
    }

    // bias + ELU
    float h[4][8];
#pragma unroll
    for (int nt = 0; nt < 4; ++nt)
#pragma unroll
      for (int e = 0; e < 8; ++e) h[nt][e] = elu1(acc[nt][e] + vb1[nt]);

    // LayerNorm per row (row M = e + hi); 16-lane butterflies give row sums
#pragma unroll
    for (int e = 0; e < 8; ++e) {
      float su = h[0][e] + h[1][e] + h[2][e] + h[3][e];
      float sq = h[0][e] * h[0][e] + h[1][e] * h[1][e] +
                 h[2][e] * h[2][e] + h[3][e] * h[3][e];
#pragma unroll
      for (int m = 1; m < 16; m <<= 1) {
        su += __shfl_xor(su, m, 32);
        sq += __shfl_xor(sq, m, 32);
      }
      float mean = su * 0.015625f;
      float var  = sq * 0.015625f - mean * mean;
      float inv  = rsqrtf(var + 1e-5f);
#pragma unroll
      for (int nt = 0; nt < 4; ++nt)
        h[nt][e] = (h[nt][e] - mean) * inv * vg[nt] + vb[nt];
    }

    // Re-stage normalized h as row-major f16 [16][64] for GEMM2 A-fragments
#pragma unroll
    for (int nt = 0; nt < 4; ++nt)
#pragma unroll
      for (int e = 0; e < 8; ++e)
        hrow[(e + hi) * 64 + nt * 16 + lo] = (_Float16)h[nt][e];

    // ---- GEMM2: [16 x 64] x [64 x 64] ----
    v8f pacc[4];
#pragma unroll
    for (int nt = 0; nt < 4; ++nt)
#pragma unroll
      for (int i = 0; i < 8; ++i) pacc[nt][i] = 0.f;

#pragma unroll
    for (int ks = 0; ks < 2; ++ks) {
      v16h a;
      const _Float16* pr = hrow + lo * 64 + ks * 32 + hi;
#pragma unroll
      for (int i = 0; i < 8; ++i) { a[i] = pr[i]; a[i + 8] = pr[i + 16]; }
#pragma unroll
      for (int nt = 0; nt < 4; ++nt)
        pacc[nt] = wmma_f16(a, *(const v16h*)sB2[ks][nt][lane], pacc[nt]);
    }

    if (PASS == 1) {
#pragma unroll
      for (int nt = 0; nt < 4; ++nt) {
        float ls = 0.f, lq = 0.f;
#pragma unroll
        for (int e = 0; e < 8; ++e) {
          float pv = elu1(pacc[nt][e] + vb2[nt]);
          ls += pv; lq += pv * pv;
        }
        csum[nt] += ls; csq[nt] += lq;
      }
    } else {
      int dsh[8];
#pragma unroll
      for (int e = 0; e < 8; ++e) dsh[e] = __shfl(didx, e + hi, 32);
#pragma unroll
      for (int nt = 0; nt < 4; ++nt) {
        const int n = nt * 16 + lo;
#pragma unroll
        for (int e = 0; e < 8; ++e) {
          float pv = elu1(pacc[nt][e] + vb2[nt]) * vscl[nt] + vsft[nt];
          atomicAdd(nodeacc + (size_t)dsh[e] * 64 + n, pv);
        }
      }
    }
  }

  if (PASS == 1) {
#pragma unroll
    for (int nt = 0; nt < 4; ++nt) {
      csum[nt] += __shfl_xor(csum[nt], 16, 32);
      csq[nt]  += __shfl_xor(csq[nt], 16, 32);
    }
    if (lane < 16) {
#pragma unroll
      for (int nt = 0; nt < 4; ++nt) {
        atomicAdd(&sBn[nt * 16 + lane], csum[nt]);
        atomicAdd(&sBn[64 + nt * 16 + lane], csq[nt]);
      }
    }
    __syncthreads();
    if (threadIdx.x < 64)
      atomicAdd(bn_sum + t * 64 + threadIdx.x, sBn[threadIdx.x]);
    else if (threadIdx.x < 128)
      atomicAdd(bn_sumsq + t * 64 + (threadIdx.x - 64), sBn[threadIdx.x]);
  }
}

// out = nodeacc @ Wr + br + feat, [N x 64] x [64 x 64]
__launch_bounds__(256)
__global__ void final_gemm_kernel(const float* __restrict__ nodeacc,
                                  const float* __restrict__ Wr,
                                  const float* __restrict__ br,
                                  const float* __restrict__ feat,
                                  float* __restrict__ out, int N) {
  __shared__ __align__(32) _Float16 sBr[2][4][32][16];
  const int lane = threadIdx.x & 31;
  const int wave = threadIdx.x >> 5;
  const int lo   = lane & 15;
  const int hi   = (lane >> 4) << 3;

  for (int idx = threadIdx.x; idx < 2 * 4 * 32; idx += 256) {
    int ks = idx >> 7, nt = (idx >> 5) & 3, l = idx & 31;
    int n = nt * 16 + (l & 15);
    int kh = (l >> 4) << 3;
#pragma unroll
    for (int e = 0; e < 16; ++e) {
      int k = ks * 32 + (e & 7) + ((e >> 3) << 4) + kh;
      sBr[ks][nt][l][e] = (_Float16)Wr[k * 64 + n];
    }
  }
  __syncthreads();

  float vbr[4];
#pragma unroll
  for (int nt = 0; nt < 4; ++nt) vbr[nt] = br[nt * 16 + lo];

  const int ntiles = (N + 15) >> 4;
  for (int tile = blockIdx.x * 8 + wave; tile < ntiles; tile += gridDim.x << 3) {
    const int base = tile << 4;
    int row = base + lo; if (row >= N) row = N - 1;
    const float* arow = nodeacc + (size_t)row * 64;

    v8f acc[4];
#pragma unroll
    for (int nt = 0; nt < 4; ++nt)
#pragma unroll
      for (int i = 0; i < 8; ++i) acc[nt][i] = 0.f;

#pragma unroll
    for (int ks = 0; ks < 2; ++ks) {
      v16h a = a_frag_f32(arow + ks * 32, hi);
#pragma unroll
      for (int nt = 0; nt < 4; ++nt)
        acc[nt] = wmma_f16(a, *(const v16h*)sBr[ks][nt][lane], acc[nt]);
    }
#pragma unroll
    for (int nt = 0; nt < 4; ++nt) {
      const int n = nt * 16 + lo;
#pragma unroll
      for (int e = 0; e < 8; ++e) {
        int r = base + e + hi;
        if (r < N)
          out[(size_t)r * 64 + n] = acc[nt][e] + vbr[nt] + feat[(size_t)r * 64 + n];
      }
    }
  }
}

extern "C" void kernel_launch(void* const* d_in, const int* in_sizes, int n_in,
                              void* d_out, int out_size, void* d_ws, size_t ws_size,
                              hipStream_t stream) {
  const float* feat = (const float*)d_in[0];
  const float* W1   = (const float*)d_in[1];
  const float* b1   = (const float*)d_in[2];
  const float* lng  = (const float*)d_in[3];
  const float* lnb  = (const float*)d_in[4];
  const float* W2   = (const float*)d_in[5];
  const float* b2   = (const float*)d_in[6];
  const float* bng  = (const float*)d_in[7];
  const float* bnb  = (const float*)d_in[8];
  const float* Wr   = (const float*)d_in[9];
  const float* br   = (const float*)d_in[10];
  const int* esrc   = (const int*)d_in[11];  // JAX default x64-off => int32
  const int* edst   = (const int*)d_in[12];
  float* out        = (float*)d_out;

  const int N = in_sizes[0] / 64;
  const int T = in_sizes[2] / 64;          // b1 is [T,64]
  const int E = in_sizes[11] / T;          // edge_src is [T,E]

  float* ws       = (float*)d_ws;
  float* bn_sum   = ws;
  float* bn_sumsq = bn_sum   + (size_t)T * 64;
  float* bn_scale = bn_sumsq + (size_t)T * 64;
  float* bn_shift = bn_scale + (size_t)T * 64;
  float* nodeacc  = bn_shift + (size_t)T * 64;

  size_t ztot = (size_t)T * 256 + (size_t)N * 64;
  zero_ws_kernel<<<512, 256, 0, stream>>>(ws, ztot);

  int ntiles = E >> 4;
  int gx = (ntiles + 7) / 8; if (gx > 960) gx = 960;
  dim3 grid(gx, T);
  edge_mlp_pass<1><<<grid, 256, 0, stream>>>(feat, W1, b1, lng, lnb, W2, b2,
      esrc, edst, bn_sum, bn_sumsq, bn_scale, bn_shift, nodeacc, E);
  bn_finalize_kernel<<<(T * 64 + 255) / 256, 256, 0, stream>>>(
      bn_sum, bn_sumsq, bng, bnb, bn_scale, bn_shift, T * 64, 1.0f / (float)E);
  edge_mlp_pass<2><<<grid, 256, 0, stream>>>(feat, W1, b1, lng, lnb, W2, b2,
      esrc, edst, bn_sum, bn_sumsq, bn_scale, bn_shift, nodeacc, E);

  int ntiles3 = (N + 15) / 16;
  int gx3 = (ntiles3 + 7) / 8; if (gx3 > 640) gx3 = 640;
  final_gemm_kernel<<<gx3, 256, 0, stream>>>(nodeacc, Wr, br, feat, out, N);
}
